// BitNetFFN_58102317580688
// MI455X (gfx1250) — 8-bit matrix operand order repaired and checked by reading, compiled, not run on hardware
//
#include <hip/hip_runtime.h>
#include <hip/hip_bf16.h>
#include <stdint.h>


typedef __attribute__((ext_vector_type(8))) int v8i;

#define D_MODEL 2048
#define D_FF    8192

__device__ inline float toF(float v) { return v; }
__device__ inline float toF(__hip_bfloat16 v) { return __bfloat162float(v); }

__device__ inline void storeOut(float* p, float v) { *p = v; }
__device__ inline void storeOut(__hip_bfloat16* p, float v) { *p = __float2bfloat16(v); }

__device__ inline void asyncLoadB128(uint32_t ldsAddr, const void* gaddr) {
    asm volatile("global_load_async_to_lds_b128 %0, %1, off"
                 :: "v"(ldsAddr), "v"(gaddr) : "memory");
}
__device__ inline void waitAsync0() {
    asm volatile("s_wait_asynccnt 0x0" ::: "memory");
}
__device__ inline uint32_t ldsOff(const void* p) {
    return (uint32_t)(uintptr_t)p;
}

__global__ void init_accum(float* accum) {
    if (threadIdx.x < 2) accum[threadIdx.x] = 0.0f;
}

__global__ __launch_bounds__(256) void abssum_kernel(const float* __restrict__ w,
                                                     size_t n, float* accum) {
    __shared__ float red[256];
    float s = 0.0f;
    for (size_t i = (size_t)blockIdx.x * 256u + threadIdx.x; i < n;
         i += (size_t)gridDim.x * 256u)
        s += fabsf(w[i]);
    red[threadIdx.x] = s;
    __syncthreads();
    for (int k = 128; k > 0; k >>= 1) {
        if ((int)threadIdx.x < k) red[threadIdx.x] += red[threadIdx.x + k];
        __syncthreads();
    }
    if (threadIdx.x == 0) atomicAdd(accum, red[0]);
}

__global__ void finalize_wdeq(const float* accum, float* wdeq,
                              float inv_n0, float inv_n1) {
    if (threadIdx.x == 0) {
        wdeq[0] = fmaxf(accum[0] * inv_n0, 1e-5f);
        wdeq[1] = fmaxf(accum[1] * inv_n1, 1e-5f);
    }
}

__global__ __launch_bounds__(256) void wquant_kernel(const float* __restrict__ w,
                                                     const float* __restrict__ wdeq,
                                                     int8_t* __restrict__ q, size_t n) {
    const float scale = 1.0f / wdeq[0];
    for (size_t i = (size_t)blockIdx.x * 256u + threadIdx.x; i < n;
         i += (size_t)gridDim.x * 256u) {
        int v = __float2int_rn(w[i] * scale);
        v = v > 1 ? 1 : (v < -1 ? -1 : v);
        q[i] = (int8_t)v;
    }
}

template <typename InT>
__global__ __launch_bounds__(256) void actquant_kernel(const InT* __restrict__ X,
                                                       const float* __restrict__ g,
                                                       int8_t* __restrict__ Q,
                                                       float* __restrict__ rowDeq,
                                                       int K) {
    __shared__ float s_norm[D_FF];
    __shared__ float s_red[256];
    const int row = blockIdx.x;
    const int tid = threadIdx.x;
    const InT* xr = X + (size_t)row * K;

    float ss = 0.0f;
    for (int i = tid; i < K; i += 256) {
        float v = toF(xr[i]);
        ss += v * v;
    }
    s_red[tid] = ss;
    __syncthreads();
    for (int k = 128; k > 0; k >>= 1) {
        if (tid < k) s_red[tid] += s_red[tid + k];
        __syncthreads();
    }
    const float rinv = 1.0f / sqrtf(s_red[0] / (float)K + 1e-6f);
    __syncthreads();

    float amax = 0.0f;
    for (int i = tid; i < K; i += 256) {
        float v = toF(xr[i]) * rinv * g[i];
        s_norm[i] = v;
        amax = fmaxf(amax, fabsf(v));
    }
    s_red[tid] = amax;
    __syncthreads();
    for (int k = 128; k > 0; k >>= 1) {
        if (tid < k) s_red[tid] = fmaxf(s_red[tid], s_red[tid + k]);
        __syncthreads();
    }
    const float mx = fmaxf(s_red[0], 1e-5f);
    const float scale = 127.0f / mx;
    if (tid == 0) rowDeq[row] = mx / 127.0f;

    int8_t* qr = Q + (size_t)row * K;
    for (int i = tid; i < K; i += 256) {
        int v = __float2int_rn(s_norm[i] * scale);
        v = v > 127 ? 127 : (v < -128 ? -128 : v);
        qr[i] = (int8_t)v;
    }
}

template <bool RELU2, typename OutT>
__global__ __launch_bounds__(256) void gemm_iu8_kernel(const int8_t* __restrict__ A,
                                                       const int8_t* __restrict__ B,
                                                       const float* __restrict__ rowDeq,
                                                       const float* __restrict__ wDeq,
                                                       OutT* __restrict__ out,
                                                       int M, int N, int K) {
    __shared__ __align__(16) int8_t As[2][64 * 64];
    __shared__ __align__(16) int8_t Bs[2][128 * 64];

    const int tid    = threadIdx.x;
    const int lane   = tid & 31;
    const int wid    = tid >> 5;
    const int laneHi = lane >> 4;
    const int lane15 = lane & 15;

    const int blockN = blockIdx.x * 128;
    const int blockM = blockIdx.y * 64;
    const int waveM  = (wid >> 2) * 32;
    const int waveN  = (wid & 3) * 32;

    v8i acc[2][2] = {};

    const int aRow = tid >> 2;
    const int aCol = (tid & 3) * 16;
    const int bRow = tid >> 1;
    const int bCol = (tid & 1) * 32;

    const int8_t* gA = A + (size_t)(blockM + aRow) * K + aCol;
    const int8_t* gB = B + (size_t)(blockN + bRow) * K + bCol;

    auto issueTile = [&](int k0, int buf) {
        asyncLoadB128(ldsOff(&As[buf][aRow * 64 + aCol]),      gA + k0);
        asyncLoadB128(ldsOff(&Bs[buf][bRow * 64 + bCol]),      gB + k0);
        asyncLoadB128(ldsOff(&Bs[buf][bRow * 64 + bCol + 16]), gB + k0 + 16);
    };

    issueTile(0, 0);
    waitAsync0();
    __syncthreads();

    int buf = 0;
    for (int k0 = 0; k0 < K; k0 += 64) {
        if (k0 + 64 < K) issueTile(k0 + 64, buf ^ 1);

        v8i afrag[2], bfrag[2];
#pragma unroll
        for (int mi = 0; mi < 2; ++mi) {
            const int8_t* ap = &As[buf][(waveM + mi * 16 + lane15) * 64 + laneHi * 8];
#pragma unroll
            for (int d = 0; d < 8; ++d) {
                const int k = ((d >> 1) << 4) + ((d & 1) << 2);
                afrag[mi][d] = *(const int*)(ap + k);
            }
        }
#pragma unroll
        for (int ni = 0; ni < 2; ++ni) {
            const int8_t* bpl = &Bs[buf][(waveN + ni * 16 + lane15) * 64 + laneHi * 8];
#pragma unroll
            for (int d = 0; d < 8; ++d) {
                const int k = ((d >> 1) << 4) + ((d & 1) << 2);
                bfrag[ni][d] = *(const int*)(bpl + k);
            }
        }

#pragma unroll
        for (int mi = 0; mi < 2; ++mi)
#pragma unroll
            for (int ni = 0; ni < 2; ++ni)
                acc[mi][ni] = __builtin_amdgcn_wmma_i32_16x16x64_iu8(
                     true, afrag[mi],  true, bfrag[ni],
                    acc[mi][ni],  false,  false);

        waitAsync0();
        __syncthreads();
        buf ^= 1;
    }

    const float wdq = wDeq[0];
#pragma unroll
    for (int mi = 0; mi < 2; ++mi) {
#pragma unroll
        for (int ni = 0; ni < 2; ++ni) {
#pragma unroll
            for (int r = 0; r < 8; ++r) {
                const int m = blockM + waveM + mi * 16 + laneHi * 8 + r;
                const int n = blockN + waveN + ni * 16 + lane15;
                float v = (float)acc[mi][ni][r] * rowDeq[m] * wdq;
                if (RELU2) { v = fmaxf(v, 0.0f); v = v * v; }
                storeOut(out + (size_t)m * N + n, v);
            }
        }
    }
}

extern "C" void kernel_launch(void* const* d_in, const int* in_sizes, int n_in,
                              void* d_out, int out_size, void* d_ws, size_t ws_size,
                              hipStream_t stream) {
    (void)n_in; (void)out_size; (void)ws_size;
    const float* x  = (const float*)d_in[0];
    const float* w1 = (const float*)d_in[1];
    const float* g1 = (const float*)d_in[2];
    const float* w2 = (const float*)d_in[3];
    const float* g2 = (const float*)d_in[4];
    float* out = (float*)d_out;

    const int M = in_sizes[0] / D_MODEL;
    const size_t NW = (size_t)D_FF * D_MODEL;

    uint8_t* ws = (uint8_t*)d_ws;
    float* accum   = (float*)ws;
    float* wdeq    = (float*)(ws + 64);
    float* rowDeq1 = (float*)(ws + 256);
    float* rowDeq2 = (float*)(ws + 256 + 4 * 8192);
    size_t off = 256 + 8 * 8192;
    int8_t* w1q = (int8_t*)(ws + off); off += NW;
    int8_t* w2q = (int8_t*)(ws + off); off += NW;
    int8_t* xq  = (int8_t*)(ws + off); off += (size_t)M * D_MODEL;
    int8_t* hq  = (int8_t*)(ws + off); off += (size_t)M * D_FF;
    __hip_bfloat16* h = (__hip_bfloat16*)(ws + off);

    init_accum<<<1, 32, 0, stream>>>(accum);
    abssum_kernel<<<1024, 256, 0, stream>>>(w1, NW, accum + 0);
    abssum_kernel<<<1024, 256, 0, stream>>>(w2, NW, accum + 1);
    finalize_wdeq<<<1, 32, 0, stream>>>(accum, wdeq, 1.0f / (float)NW, 1.0f / (float)NW);

    wquant_kernel<<<4096, 256, 0, stream>>>(w1, wdeq + 0, w1q, NW);
    wquant_kernel<<<4096, 256, 0, stream>>>(w2, wdeq + 1, w2q, NW);

    actquant_kernel<float><<<M, 256, 0, stream>>>(x, g1, xq, rowDeq1, D_MODEL);
    gemm_iu8_kernel<true, __hip_bfloat16>
        <<<dim3(D_FF / 128, M / 64), 256, 0, stream>>>(
            xq, w1q, rowDeq1, wdeq + 0, h, M, D_FF, D_MODEL);

    actquant_kernel<__hip_bfloat16><<<M, 256, 0, stream>>>(h, g2, hq, rowDeq2, D_FF);
    gemm_iu8_kernel<false, float>
        <<<dim3(D_MODEL / 128, M / 64), 256, 0, stream>>>(
            hq, w2q, rowDeq2, wdeq + 1, out, M, D_MODEL, D_FF);
}
